// MultiHeadSelfAttention_30743375905122
// MI455X (gfx1250) — matrix operand order repaired and checked by reading, not compiled, not run on hardware
//
#include <hip/hip_runtime.h>

typedef __attribute__((ext_vector_type(16))) _Float16 v16h;
typedef __attribute__((ext_vector_type(8)))  _Float16 h8;
typedef __attribute__((ext_vector_type(8)))  float    v8f;

#define HIDDEN 1024
#define HEADS  16
#define HD     64
#define BATCH  4
#define SEQ    2048
#define MTOT   (BATCH * SEQ)   // 8192 tokens

union ABFrag { v16h v; h8 h[2]; };

static __device__ __forceinline__ v16h load_a16(const _Float16* p0, const _Float16* p1) {
  ABFrag u;
  u.h[0] = *(const h8*)p0;         // lanes 0-15: K 0..7  | lanes 16-31: K 16..23
  u.h[1] = *(const h8*)(p0 + 8);   // lanes 0-15: K 8..15 | lanes 16-31: K 24..31
  return u.v;
}
static __device__ __forceinline__ v16h load_b16(const _Float16* p) {
  return *(const v16h*)p;    // per-lane contiguous 16 K-values of one B column
}
static __device__ __forceinline__ v8f wmma_f16(v16h a, v16h b, v8f c) {
  return __builtin_amdgcn_wmma_f32_16x16x32_f16(false, a, false, b, (short)0, c, false, false);
}

// Raw v_exp_f32: args here are always <= 0, underflow-to-zero is exactly what
// online softmax wants, so skip libm's |x|>126 range fixup (cndmask+ldexp).
static __device__ __forceinline__ float fast_exp2(float x) {
  return __builtin_amdgcn_exp2f(x);
}

// ---- DPP16 butterfly reduction over the 16-lane half-groups (no LDS, no waits)
// 0xB1 = quad_perm[1,0,3,2] (xor1), 0x4E = quad_perm[2,3,0,1] (xor2),
// 0x141 = row_half_mirror (== xor4 once quads are uniform),
// 0x140 = row_mirror      (== xor8 once 8-groups are uniform).
template<int CTRL>
static __device__ __forceinline__ float dpp_bcast(float x) {
  int r = __builtin_amdgcn_update_dpp(0, __builtin_bit_cast(int, x),
                                      CTRL, 0xF, 0xF, true);
  return __builtin_bit_cast(float, r);
}
static __device__ __forceinline__ float rowmax16(float x) {
  x = fmaxf(x, dpp_bcast<0xB1>(x));
  x = fmaxf(x, dpp_bcast<0x4E>(x));
  x = fmaxf(x, dpp_bcast<0x141>(x));
  x = fmaxf(x, dpp_bcast<0x140>(x));
  return x;
}
static __device__ __forceinline__ float rowsum16(float x) {
  x += dpp_bcast<0xB1>(x);
  x += dpp_bcast<0x4E>(x);
  x += dpp_bcast<0x141>(x);
  x += dpp_bcast<0x140>(x);
  return x;
}

// ---------------------------------------------------------------- cast fp32 -> f16
__global__ __launch_bounds__(256) void cast_kernel(const float* __restrict__ s,
                                                   _Float16* __restrict__ d, int n8) {
  int i = blockIdx.x * blockDim.x + threadIdx.x;
  if (i < n8) {
    v8f x = ((const v8f*)s)[i];
    h8 y;
#pragma unroll
    for (int j = 0; j < 8; ++j) y[j] = (_Float16)x[j];
    ((h8*)d)[i] = y;
  }
}

// ---------------------------------------------------------------- projection GEMM
// out = (x @ W^T + b) * oscale
// TRANSPOSE=0 -> [B,H,S,64] (Q/K), TRANSPOSE=1 -> [B,H,64,S] (V^T)
// A/B fragments explicitly double-buffered so step k+1 loads overlap step k WMMAs.
template<int TRANSPOSE>
__global__ __launch_bounds__(256) void proj_kernel(
    const _Float16* __restrict__ xh, const _Float16* __restrict__ w,
    const float* __restrict__ bias, _Float16* __restrict__ out, float oscale)
{
  const int lane = threadIdx.x & 31;
  const int wave = blockIdx.x * 8 + (threadIdx.x >> 5);
  const int mt = wave >> 4;          // 512 M-tiles of 16 tokens
  const int ns = wave & 15;          // 16 N-strips of 64 (= one head)
  const int m0 = mt << 4;
  const int n0 = ns << 6;
  const int lhi = lane >> 4, llo = lane & 15;

  const _Float16* arow = xh + (size_t)(m0 + llo) * HIDDEN + lhi * 16;
  const _Float16* brow[4];
#pragma unroll
  for (int t = 0; t < 4; ++t)
    brow[t] = w + (size_t)(n0 + t * 16 + llo) * HIDDEN + lhi * 16;

  v8f acc[4] = {};
  v16h a = load_a16(arow, arow + 16);
  v16h b[4];
#pragma unroll
  for (int t = 0; t < 4; ++t) b[t] = load_b16(brow[t]);

  for (int k0 = 0; k0 < HIDDEN - 32; k0 += 32) {
    const int kn = k0 + 32;
    v16h an = load_a16(arow + kn, arow + kn + 16);
    v16h bn[4];
#pragma unroll
    for (int t = 0; t < 4; ++t) bn[t] = load_b16(brow[t] + kn);
#pragma unroll
    for (int t = 0; t < 4; ++t) acc[t] = wmma_f16(a, b[t], acc[t]);
    a = an;
#pragma unroll
    for (int t = 0; t < 4; ++t) b[t] = bn[t];
  }
#pragma unroll
  for (int t = 0; t < 4; ++t) acc[t] = wmma_f16(a, b[t], acc[t]);

  const int bb_ = m0 >> 11;                 // batch (2048 tokens each)
  const int s0  = (m0 & (SEQ - 1)) + lhi * 8;
  const int h   = ns;
#pragma unroll
  for (int t = 0; t < 4; ++t) {
    const int d = t * 16 + llo;
    const float bv = bias[n0 + t * 16 + llo];
    if (TRANSPOSE == 0) {
      _Float16* p = out + ((size_t)(bb_ * HEADS + h) * SEQ + s0) * HD + d;
#pragma unroll
      for (int v = 0; v < 8; ++v)
        p[(size_t)v * HD] = (_Float16)((acc[t][v] + bv) * oscale);
    } else {
      h8 vv;
#pragma unroll
      for (int v = 0; v < 8; ++v) vv[v] = (_Float16)((acc[t][v] + bv) * oscale);
      *(h8*)(out + ((size_t)(bb_ * HEADS + h) * HD + d) * SEQ + s0) = vv;
    }
  }
}

// ---------------------------------------------------------------- flash attention
// Q was pre-scaled by (1/sqrt(64))*log2(e), so scores are already in the
// log2 domain: p = exp2(s - m), alpha = exp2(m_old - m_new).
__global__ __launch_bounds__(128) void attn_kernel(
    const _Float16* __restrict__ Q, const _Float16* __restrict__ K,
    const _Float16* __restrict__ Vt, const int* __restrict__ mask,
    _Float16* __restrict__ Ao)
{
  __shared__ __align__(32) _Float16 pbuf[4][16 * 64];  // per-wave P tile (f16)
  const int lane = threadIdx.x & 31;
  const int w    = threadIdx.x >> 5;
  const int bh   = blockIdx.x >> 5;        // (b*16 + h)
  const int qc   = blockIdx.x & 31;        // 32 q-chunks of 64
  const int b = bh >> 4, h = bh & 15;
  const int q0 = qc * 64 + w * 16;
  const int lhi = lane >> 4, llo = lane & 15;

  const _Float16* Qb = Q  + (size_t)bh * SEQ * HD;
  const _Float16* Kb = K  + (size_t)bh * SEQ * HD;
  const _Float16* Vb = Vt + (size_t)bh * HD * SEQ;
  const int* mb = mask + b * SEQ;

  const _Float16* qrow = Qb + (size_t)(q0 + llo) * HD + lhi * 16;
  const v16h aq0 = load_a16(qrow,      qrow + 16);   // d 0..31
  const v16h aq1 = load_a16(qrow + 32, qrow + 48);   // d 32..63

  float mi[8], li[8];
#pragma unroll
  for (int v = 0; v < 8; ++v) { mi[v] = -1e30f; li[v] = 0.f; }
  v8f acc[4] = {};

  _Float16* pb = pbuf[w];

  // preload K B-fragments for the first key chunk
  v16h kb[8];
#pragma unroll
  for (int t = 0; t < 4; ++t) {
    const _Float16* krow = Kb + (size_t)(t * 16 + llo) * HD + lhi * 16;
    kb[2 * t]     = load_b16(krow);
    kb[2 * t + 1] = load_b16(krow + 32);
  }

  for (int kc = 0; kc < SEQ; kc += 64) {
    // -------- scores S = Q K^T (16 x 64 chunk), K frags already resident
    v8f sc[4] = {};
#pragma unroll
    for (int t = 0; t < 4; ++t) {
      sc[t] = wmma_f16(aq0, kb[2 * t],     sc[t]);
      sc[t] = wmma_f16(aq1, kb[2 * t + 1], sc[t]);
    }

    // -------- issue V loads + next-chunk K loads; softmax math hides them
    v16h vb[8];
#pragma unroll
    for (int t = 0; t < 4; ++t) {
      const _Float16* vrow = Vb + (size_t)(t * 16 + llo) * SEQ + kc + lhi * 16;
      vb[2 * t]     = load_b16(vrow);
      vb[2 * t + 1] = load_b16(vrow + 32);
    }
    if (kc + 64 < SEQ) {
#pragma unroll
      for (int t = 0; t < 4; ++t) {
        const _Float16* krow = Kb + (size_t)(kc + 64 + t * 16 + llo) * HD + lhi * 16;
        __builtin_prefetch((const void*)(krow + 64 * HD), 0, 1);  // chunk after next
        kb[2 * t]     = load_b16(krow);
        kb[2 * t + 1] = load_b16(krow + 32);
      }
    }
    float mk[4];
#pragma unroll
    for (int t = 0; t < 4; ++t)
      mk[t] = (mb[kc + t * 16 + llo] == 0) ? -1e9f : 0.0f;
#pragma unroll
    for (int t = 0; t < 4; ++t)
#pragma unroll
      for (int v = 0; v < 8; ++v)
        sc[t][v] += mk[t];

    // -------- online softmax: DPP16 butterfly reductions (pure VALU)
    float alpha[8];
#pragma unroll
    for (int v = 0; v < 8; ++v) {
      float x = fmaxf(fmaxf(sc[0][v], sc[1][v]), fmaxf(sc[2][v], sc[3][v]));
      x = rowmax16(x);
      const float mn = fmaxf(mi[v], x);
      alpha[v] = fast_exp2(mi[v] - mn);   // arg <= 0: raw v_exp_f32 is safe
      mi[v] = mn;
    }
    float rs[8];
#pragma unroll
    for (int v = 0; v < 8; ++v) rs[v] = 0.f;
#pragma unroll
    for (int t = 0; t < 4; ++t)
#pragma unroll
      for (int v = 0; v < 8; ++v) {
        const float p = fast_exp2(sc[t][v] - mi[v]);   // arg <= 0
        rs[v] += p;
        pb[(v + 8 * lhi) * 64 + t * 16 + llo] = (_Float16)p;  // D-layout -> row-major
      }
#pragma unroll
    for (int v = 0; v < 8; ++v)
      li[v] = li[v] * alpha[v] + rowsum16(rs[v]);
#pragma unroll
    for (int t = 0; t < 4; ++t)
#pragma unroll
      for (int v = 0; v < 8; ++v) acc[t][v] *= alpha[v];

    __syncthreads();   // P tile visible (LDS)
    const _Float16* prow = pb + llo * 64 + lhi * 16;
    const v16h ap0 = load_a16(prow,      prow + 16);   // keys kc+0..31
    const v16h ap1 = load_a16(prow + 32, prow + 48);   // keys kc+32..63
#pragma unroll
    for (int t = 0; t < 4; ++t) {
      acc[t] = wmma_f16(ap0, vb[2 * t],     acc[t]);
      acc[t] = wmma_f16(ap1, vb[2 * t + 1], acc[t]);
    }
    __syncthreads();   // done reading pb before next overwrite
  }

  // -------- epilogue: O /= l, store [token, hidden] f16
#pragma unroll
  for (int v = 0; v < 8; ++v) li[v] = __builtin_amdgcn_rcpf(li[v]);
#pragma unroll
  for (int t = 0; t < 4; ++t) {
    const int col = h * HD + t * 16 + llo;
#pragma unroll
    for (int v = 0; v < 8; ++v) {
      const int tok = b * SEQ + q0 + v + 8 * lhi;
      Ao[(size_t)tok * HIDDEN + col] = (_Float16)(acc[t][v] * li[v]);
    }
  }
}

// ---------------------------------------------------------------- output projection (f32 out)
__global__ __launch_bounds__(256) void oproj_kernel(
    const _Float16* __restrict__ A, const _Float16* __restrict__ wo,
    const float* __restrict__ bo, float* __restrict__ out)
{
  const int lane = threadIdx.x & 31;
  const int wave = blockIdx.x * 8 + (threadIdx.x >> 5);
  const int mt = wave >> 4, ns = wave & 15;
  const int m0 = mt << 4, n0 = ns << 6;
  const int lhi = lane >> 4, llo = lane & 15;

  const _Float16* arow = A + (size_t)(m0 + llo) * HIDDEN + lhi * 16;
  const _Float16* brow[4];
#pragma unroll
  for (int t = 0; t < 4; ++t)
    brow[t] = wo + (size_t)(n0 + t * 16 + llo) * HIDDEN + lhi * 16;

  v8f acc[4] = {};
  v16h a = load_a16(arow, arow + 16);
  v16h b[4];
#pragma unroll
  for (int t = 0; t < 4; ++t) b[t] = load_b16(brow[t]);

  for (int k0 = 0; k0 < HIDDEN - 32; k0 += 32) {
    const int kn = k0 + 32;
    v16h an = load_a16(arow + kn, arow + kn + 16);
    v16h bn[4];
#pragma unroll
    for (int t = 0; t < 4; ++t) bn[t] = load_b16(brow[t] + kn);
#pragma unroll
    for (int t = 0; t < 4; ++t) acc[t] = wmma_f16(a, b[t], acc[t]);
    a = an;
#pragma unroll
    for (int t = 0; t < 4; ++t) b[t] = bn[t];
  }
#pragma unroll
  for (int t = 0; t < 4; ++t) acc[t] = wmma_f16(a, b[t], acc[t]);

#pragma unroll
  for (int t = 0; t < 4; ++t) {
    const int n = n0 + t * 16 + llo;
    const float bb = bo[n];
#pragma unroll
    for (int v = 0; v < 8; ++v) {
      const int m = m0 + v + 8 * lhi;
      out[(size_t)m * HIDDEN + n] = acc[t][v] + bb;
    }
  }
}

// ---------------------------------------------------------------- host entry
extern "C" void kernel_launch(void* const* d_in, const int* in_sizes, int n_in,
                              void* d_out, int out_size, void* d_ws, size_t ws_size,
                              hipStream_t stream)
{
  (void)in_sizes; (void)n_in; (void)out_size; (void)ws_size;
  const float* x  = (const float*)d_in[0];
  const int* mask = (const int*)d_in[1];
  const float* Wq = (const float*)d_in[2];
  const float* bq = (const float*)d_in[3];
  const float* Wk = (const float*)d_in[4];
  const float* bk = (const float*)d_in[5];
  const float* Wv = (const float*)d_in[6];
  const float* bv = (const float*)d_in[7];
  const float* Wo = (const float*)d_in[8];
  const float* bo = (const float*)d_in[9];
  float* out = (float*)d_out;

  char* ws = (char*)d_ws;                       // layout (MB offsets):
  _Float16* xh  = (_Float16*)(ws);              //  0: x f16      (16 MB)
  _Float16* wqh = (_Float16*)(ws + (16u << 20));// 16: Wq f16     ( 2 MB)
  _Float16* wkh = (_Float16*)(ws + (18u << 20));
  _Float16* wvh = (_Float16*)(ws + (20u << 20));
  _Float16* woh = (_Float16*)(ws + (22u << 20));
  _Float16* Qb  = (_Float16*)(ws + (24u << 20));// 24: Q [B,H,S,64]   (16 MB)
  _Float16* Kb  = (_Float16*)(ws + (40u << 20));// 40: K [B,H,S,64]   (16 MB)
  _Float16* Vtb = (_Float16*)(ws + (56u << 20));// 56: V^T [B,H,64,S] (16 MB)
  _Float16* Ab  = (_Float16*)(ws + (72u << 20));// 72: attn out f16   (16 MB)

  const int nx8 = MTOT * HIDDEN / 8;
  const int nw8 = HIDDEN * HIDDEN / 8;
  cast_kernel<<<(nx8 + 255) / 256, 256, 0, stream>>>(x,  xh,  nx8);
  cast_kernel<<<(nw8 + 255) / 256, 256, 0, stream>>>(Wq, wqh, nw8);
  cast_kernel<<<(nw8 + 255) / 256, 256, 0, stream>>>(Wk, wkh, nw8);
  cast_kernel<<<(nw8 + 255) / 256, 256, 0, stream>>>(Wv, wvh, nw8);
  cast_kernel<<<(nw8 + 255) / 256, 256, 0, stream>>>(Wo, woh, nw8);

  // Q is pre-scaled by (1/sqrt(HD)) * log2(e) so attention works in log2 domain
  const float qscale = 0.125f * 1.44269504f;
  proj_kernel<0><<<1024, 256, 0, stream>>>(xh, wqh, bq, Qb, qscale);
  proj_kernel<0><<<1024, 256, 0, stream>>>(xh, wkh, bk, Kb, 1.0f);
  proj_kernel<1><<<1024, 256, 0, stream>>>(xh, wvh, bv, Vtb, 1.0f);

  // one block = 4 q-tiles of one (b,h); 64 bh x 32 chunks
  attn_kernel<<<BATCH * HEADS * (SEQ / 64), 128, 0, stream>>>(Qb, Kb, Vtb, mask, Ab);

  oproj_kernel<<<1024, 256, 0, stream>>>(Ab, woh, bo, out);
}
